// GNNModel_29463475650682
// MI455X (gfx1250) — matrix operand order repaired and checked by reading, not compiled, not run on hardware
//
#include <hip/hip_runtime.h>

#define NN 50000
#define NE 800000

typedef __attribute__((ext_vector_type(16))) __bf16 v16bf;
typedef __attribute__((ext_vector_type(8)))  float  v8f;

union BFrag { v16bf v; unsigned short u[16]; unsigned int w[8]; };

// RNE f32->bf16: used only in the one-shot weight pack (off hot path).
__device__ __forceinline__ unsigned short f2bf_rne(float f){
  unsigned int x = __float_as_uint(f);
  x += 0x7FFFu + ((x >> 16) & 1u);
  return (unsigned short)(x >> 16);
}

// Truncating f32->bf16: compiles to a D16_HI store (conversion is free in the store).
__device__ __forceinline__ unsigned short f2bf(float f){
  return (unsigned short)(__float_as_uint(f) >> 16);
}

// A fragment (16x32 bf16) from LDS tile, row stride `stride` (ushorts), K offset kOff.
// ISA 7.12.2: lanes 0-15 -> M=lane, VGPR t pairs cover K {0..7,16..23};
// lanes 16-31 same M, K {8..15,24..31}.  -> two 16B contiguous chunks per lane.
__device__ __forceinline__ void loadA_lds(const unsigned short* tile, int stride, int kOff, BFrag& a){
  const int lane = threadIdx.x & 31;
  const int m = lane & 15;
  const int half = lane >> 4;
  const unsigned short* p = tile + m * stride + kOff + half * 8;
#pragma unroll
  for (int t = 0; t < 8; ++t){
    const int k = ((t < 4) ? 0 : 16) + (t & 3) * 2;
    a.u[2*t]   = p[k];
    a.u[2*t+1] = p[k+1];
  }
}

// B fragment (32x16 bf16) from pre-packed bf16 weights Wp[dout][dinPad] (row-major = B^T).
// lane -> column n = (lane&15)+16*nBlk ; VGPR t -> K pair (kOff + 8*half + 2t + (t>=4 ? 8 : 0)).
// two 16B chunks (K +0..7, +16..23) per lane -> two b128 loads, no conversion VALU.
__device__ __forceinline__ void loadB_pk(const unsigned short* Wp, int dinPad, int nBlk, int kOff, BFrag& b){
  const int lane = threadIdx.x & 31;
  const int n = (lane & 15) + nBlk * 16;
  const int half = lane >> 4;
  const unsigned int* p = (const unsigned int*)(Wp + (size_t)n * dinPad + kOff + half * 8);
#pragma unroll
  for (int t = 0; t < 8; ++t) b.w[t] = p[(t < 4) ? t : t + 4];
}

// One dense layer on a 16-row LDS tile: out = act(in @ W^T + bias), bf16 in/out, f32 accum.
template<int DIN, int DOUT, int ACT>
__device__ __forceinline__ void mlp_layer(const unsigned short* inT, int inS,
                                          unsigned short* outT, int outS,
                                          const unsigned short* Wp, const float* bias){
  constexpr int KPAD = (DIN + 31) & ~31;
  constexpr int KS = KPAD / 32;
  constexpr int NB = DOUT / 16;
  const int lane = threadIdx.x & 31;
  const int n0 = lane & 15;
  const int half = lane >> 4;
#pragma unroll
  for (int nb = 0; nb < NB; ++nb){
    v8f c = {};
#pragma unroll
    for (int ks = 0; ks < KS; ++ks){
      BFrag a, b;
      loadA_lds(inT, inS, ks * 32, a);
      loadB_pk(Wp, KPAD, nb, ks * 32, b);
      c = __builtin_amdgcn_wmma_f32_16x16x32_bf16(false, a.v, false, b.v,
                                                  (short)0, c, false, false);
    }
    const int n = n0 + nb * 16;
    const float bv = bias[n];
#pragma unroll
    for (int v = 0; v < 8; ++v){
      float f = c[v] + bv;
      if (ACT == 1) f = fmaxf(f, 0.0f);
      outT[(v + 8 * half) * outS + n] = f2bf(f);  // C layout: row = v + 8*half, col = n
    }
  }
}

// ---------------- one-shot weight pre-pack: f32 [dout][din] -> bf16 [dout][dinPad] -----
__global__ void __launch_bounds__(256)
pack_w_kernel(const float* __restrict__ W, unsigned short* __restrict__ out,
              int dout, int din, int dinPad){
  const int i = blockIdx.x * 256 + threadIdx.x;
  if (i >= dout * dinPad) return;
  const int n = i / dinPad;
  const int k = i - n * dinPad;
  out[i] = (k < din) ? f2bf_rne(W[n * din + k]) : (unsigned short)0;
}

// ---------------- node preprocessing: dummy replace + LayerNorm(bn0) ----------------
__global__ void __launch_bounds__(256)
node_pre_kernel(const float* __restrict__ xin, const float* __restrict__ dummy,
                const float* __restrict__ g, const float* __restrict__ b,
                float* __restrict__ xlnf, unsigned short* __restrict__ xlnb){
  const int wave = threadIdx.x >> 5;
  const int lane = threadIdx.x & 31;
  const long long n = (long long)blockIdx.x * 8 + wave;
  if (n >= NN) return;
  const float x0 = xin[n * 32];
  float v = (x0 == -999.0f) ? dummy[lane] : xin[n * 32 + lane];
  float s = v;
#pragma unroll
  for (int o = 16; o > 0; o >>= 1) s += __shfl_xor(s, o, 32);
  const float m = s * (1.0f / 32.0f);
  const float d = v - m;
  float q = d * d;
#pragma unroll
  for (int o = 16; o > 0; o >>= 1) q += __shfl_xor(q, o, 32);
  const float y = d * rsqrtf(q * (1.0f / 32.0f) + 1e-6f) * g[lane] + b[lane];
  xlnf[n * 32 + lane] = y;
  xlnb[n * 32 + lane] = f2bf(y);
}

// ---------------- fused edge encoder + classifier, WMMA 16-edge tiles ----------------
__global__ void __launch_bounds__(128)
edge_enc_kernel(const float* __restrict__ eattr, const long long* __restrict__ eidx,
                const float* __restrict__ lg, const float* __restrict__ lb,
                const unsigned short* __restrict__ W1p, const float* b1,
                const unsigned short* __restrict__ W2p, const float* b2,
                const unsigned short* __restrict__ W3p, const float* b3,
                const float* __restrict__ Wc1, const float* __restrict__ bc1,
                const float* __restrict__ Wc2, const float* __restrict__ bc2,
                unsigned short* __restrict__ eencb,
                float* __restrict__ efsum, float* __restrict__ deg){
  constexpr int S0 = 136, S1 = 72;          // padded strides vs 64-bank LDS
  __shared__ unsigned short buf0[4][16 * S0];
  __shared__ unsigned short buf1[4][16 * S1];
  __shared__ float wgt[4][16];
  __shared__ int   dstl[4][16];
  const int wave = threadIdx.x >> 5;
  const int lane = threadIdx.x & 31;
  unsigned short* B0 = buf0[wave];
  unsigned short* B1 = buf1[wave];
  const long long base = ((long long)blockIdx.x * 4 + wave) * 16;
  if (base >= NE) return;

  if (lane < 16){                            // stage one edge row per lane
    const long long e = base + lane;
    const bool valid = (e < NE);
    const long long ec = valid ? e : (NE - 1);
    float fa[16];
#pragma unroll
    for (int j = 0; j < 16; ++j) fa[j] = eattr[ec * 16 + j];
    float m = 0.f;
#pragma unroll
    for (int j = 0; j < 16; ++j) m += fa[j];
    m *= (1.0f / 16.0f);
    float var = 0.f;
#pragma unroll
    for (int j = 0; j < 16; ++j){ const float d = fa[j] - m; var += d * d; }
    const float inv = rsqrtf(var * (1.0f / 16.0f) + 1e-6f);
    unsigned short* row = B0 + lane * S0;
#pragma unroll
    for (int j = 0; j < 16; ++j) row[j] = f2bf((fa[j] - m) * inv * lg[j] + lb[j]);
#pragma unroll
    for (int j = 16; j < 32; ++j) row[j] = 0;  // zero-pad K 16..31

    // edge classifier on RAW e_attr: 16->16 relu, 16->1 sigmoid
    float acc = bc2[0];
#pragma unroll
    for (int k = 0; k < 16; ++k){
      float h = bc1[k];
#pragma unroll
      for (int j = 0; j < 16; ++j) h += fa[j] * Wc1[k * 16 + j];
      acc += fmaxf(h, 0.0f) * Wc2[k];
    }
    const float w = 1.0f / (1.0f + __expf(-acc));
    wgt[wave][lane] = valid ? w : 0.0f;
    const int d = valid ? (int)eidx[NE + e] : -1;
    dstl[wave][lane] = d;
    if (valid) atomicAdd(deg + d, 1.0f);
  }

  mlp_layer<16, 64, 1>(B0, S0, B1, S1, W1p, b1);    // relu
  mlp_layer<64, 128, 1>(B1, S1, B0, S0, W2p, b2);   // relu (input in B0 is dead)

  // final 128->64, scale by classifier weight, emit bf16 e_enc + ef_sum atomics
  const int n0 = lane & 15;
  const int half = lane >> 4;
#pragma unroll
  for (int nb = 0; nb < 4; ++nb){
    v8f c = {};
#pragma unroll
    for (int ks = 0; ks < 4; ++ks){
      BFrag a, b;
      loadA_lds(B0, S0, ks * 32, a);
      loadB_pk(W3p, 128, nb, ks * 32, b);
      c = __builtin_amdgcn_wmma_f32_16x16x32_bf16(false, a.v, false, b.v,
                                                  (short)0, c, false, false);
    }
    const int n = n0 + nb * 16;
    const float bv = b3[n];
#pragma unroll
    for (int v = 0; v < 8; ++v){
      const int mr = v + 8 * half;
      const int d = dstl[wave][mr];
      if (d >= 0){
        const float val = (c[v] + bv) * wgt[wave][mr];
        eencb[(base + mr) * 64 + n] = f2bf(val);
        atomicAdd(efsum + (long long)d * 64 + n, val);
      }
    }
  }
}

// ---------------- EdgeMLPConv: gather [x[dst]|x[src]|e_enc] -> 3-layer MLP -> scatter ----
template<int DOUT>
__global__ void __launch_bounds__(128)
conv_kernel(const unsigned short* __restrict__ xb,      // [N,32] bf16 node feats
            const unsigned short* __restrict__ eencb,   // [E,64] bf16 edge feats
            const long long* __restrict__ eidx,
            const unsigned short* __restrict__ W1p, const float* b1,
            const unsigned short* __restrict__ W2p, const float* b2,
            const unsigned short* __restrict__ W3p, const float* b3,
            float* __restrict__ outSum){
  constexpr int S0 = 136, S1 = 72, S2 = 72;
  __shared__ unsigned short buf0[4][16 * S0];
  __shared__ unsigned short buf1[4][16 * S1];
  __shared__ unsigned short buf2[4][16 * S2];
  __shared__ int dstl[4][16];
  const int wave = threadIdx.x >> 5;
  const int lane = threadIdx.x & 31;
  unsigned short* B0 = buf0[wave];
  unsigned short* B1 = buf1[wave];
  unsigned short* B2 = buf2[wave];
  const long long base = ((long long)blockIdx.x * 4 + wave) * 16;
  if (base >= NE) return;

  if (lane < 16){
    const long long e = base + lane;
    unsigned int* r4 = (unsigned int*)(B0 + lane * S0);  // S0 even -> dword aligned
    if (e < NE){
      const int s = (int)eidx[e];
      const int d = (int)eidx[NE + e];
      const unsigned int* xd4 = (const unsigned int*)(xb + (long long)d * 32);
      const unsigned int* xs4 = (const unsigned int*)(xb + (long long)s * 32);
      const unsigned int* ee4 = (const unsigned int*)(eencb + e * 64);
#pragma unroll
      for (int j = 0; j < 16; ++j){ r4[j] = xd4[j]; r4[16 + j] = xs4[j]; }
#pragma unroll
      for (int j = 0; j < 32; ++j) r4[32 + j] = ee4[j];
      dstl[wave][lane] = d;
    } else {
#pragma unroll
      for (int j = 0; j < 64; ++j) r4[j] = 0u;
      dstl[wave][lane] = -1;
    }
  }

  mlp_layer<128, DOUT, 1>(B0, S0, B1, S1, W1p, b1);   // relu
  mlp_layer<DOUT, DOUT, 1>(B1, S1, B2, S2, W2p, b2);  // relu

  const int n0 = lane & 15;
  const int half = lane >> 4;
#pragma unroll
  for (int nb = 0; nb < DOUT / 16; ++nb){
    v8f c = {};
#pragma unroll
    for (int ks = 0; ks < DOUT / 32; ++ks){
      BFrag a, b;
      loadA_lds(B2, S2, ks * 32, a);
      loadB_pk(W3p, DOUT, nb, ks * 32, b);
      c = __builtin_amdgcn_wmma_f32_16x16x32_bf16(false, a.v, false, b.v,
                                                  (short)0, c, false, false);
    }
    const int n = n0 + nb * 16;
    const float bv = b3[n];
#pragma unroll
    for (int v = 0; v < 8; ++v){
      const int d = dstl[wave][v + 8 * half];
      if (d >= 0) atomicAdd(outSum + (long long)d * DOUT + n, c[v] + bv);
    }
  }
}

// ---------------- mean + LN(bn1) + leaky_relu -> bf16 x1 ----------------
__global__ void __launch_bounds__(256)
node_mid_kernel(const float* __restrict__ x1sum, const float* __restrict__ deg,
                const float* __restrict__ g, const float* __restrict__ b,
                unsigned short* __restrict__ x1b){
  const int wave = threadIdx.x >> 5;
  const int lane = threadIdx.x & 31;
  const long long n = (long long)blockIdx.x * 8 + wave;
  if (n >= NN) return;
  const float dc = fmaxf(deg[n], 1.0f);
  const float v = x1sum[n * 32 + lane] / dc;
  float s = v;
#pragma unroll
  for (int o = 16; o > 0; o >>= 1) s += __shfl_xor(s, o, 32);
  const float m = s * (1.0f / 32.0f);
  const float d = v - m;
  float q = d * d;
#pragma unroll
  for (int o = 16; o > 0; o >>= 1) q += __shfl_xor(q, o, 32);
  float y = d * rsqrtf(q * (1.0f / 32.0f) + 1e-6f) * g[lane] + b[lane];
  y = (y > 0.0f) ? y : 0.01f * y;
  x1b[n * 32 + lane] = f2bf(y);
}

// ---------------- mean+LN+relu, gated skip, ef_mean, concat out, prediction head ------
__global__ void __launch_bounds__(256)
node_final_kernel(const float* __restrict__ x2sum, const float* __restrict__ deg,
                  const float* __restrict__ efsum, const float* __restrict__ xlnf,
                  const float* __restrict__ g2, const float* __restrict__ b2,
                  const float* __restrict__ Wps, const float* __restrict__ bps,
                  const float* __restrict__ Wg,  const float* __restrict__ bg,
                  const float* __restrict__ Wn1, const float* __restrict__ bn1,
                  const float* __restrict__ Wn2, const float* __restrict__ bn2,
                  const float* __restrict__ Wn3, const float* __restrict__ bn3,
                  float* __restrict__ out){
  __shared__ float sh[8][320];
  const int wave = threadIdx.x >> 5;
  const int lane = threadIdx.x & 31;
  const long long n = (long long)blockIdx.x * 8 + wave;
  if (n >= NN) return;
  float* XL = sh[wave];        // [0,32)   x_ln
  float* SK = XL + 32;         // [32,96)  skip
  float* CB = XL + 96;         // [96,224) xf_combined
  float* H1 = XL + 224;        // [224,288) head hidden 1
  const float dc = fmaxf(deg[n], 1.0f);

  // x2 = relu(LN(x2sum/deg))
  const float a0 = x2sum[n * 64 + lane] / dc;
  const float a1 = x2sum[n * 64 + 32 + lane] / dc;
  float s = a0 + a1;
#pragma unroll
  for (int o = 16; o > 0; o >>= 1) s += __shfl_xor(s, o, 32);
  const float m = s * (1.0f / 64.0f);
  float q = (a0 - m) * (a0 - m) + (a1 - m) * (a1 - m);
#pragma unroll
  for (int o = 16; o > 0; o >>= 1) q += __shfl_xor(q, o, 32);
  const float inv = rsqrtf(q * (1.0f / 64.0f) + 1e-6f);
  float x2v[2];
  x2v[0] = fmaxf((a0 - m) * inv * g2[lane]      + b2[lane],      0.0f);
  x2v[1] = fmaxf((a1 - m) * inv * g2[lane + 32] + b2[lane + 32], 0.0f);

  XL[lane] = xlnf[n * 32 + lane];

#pragma unroll
  for (int jj = 0; jj < 2; ++jj){                 // skip = x_ln @ Wps^T + bps
    const int j = lane + 32 * jj;
    float acc = bps[j];
    for (int i = 0; i < 32; ++i) acc += XL[i] * Wps[j * 32 + i];
    SK[j] = acc;
  }
#pragma unroll
  for (int jj = 0; jj < 2; ++jj){                 // gate, xf, ef_mean, combined out
    const int j = lane + 32 * jj;
    float acc = bg[j];
    for (int i = 0; i < 64; ++i) acc += SK[i] * Wg[j * 64 + i];
    const float gt = 1.0f / (1.0f + __expf(-acc));
    const float xf = gt * SK[j] + (1.0f - gt) * x2v[jj];
    CB[j] = xf;
    out[n * 128 + j] = xf;
    const float em = efsum[n * 64 + j] / dc;
    CB[64 + j] = em;
    out[n * 128 + 64 + j] = em;
  }
#pragma unroll
  for (int jj = 0; jj < 2; ++jj){                 // head: 128 -> 64, elu
    const int j = lane + 32 * jj;
    float acc = bn1[j];
    for (int i = 0; i < 128; ++i) acc += CB[i] * Wn1[j * 128 + i];
    H1[j] = (acc > 0.0f) ? acc : (__expf(acc) - 1.0f);
  }
  {                                               // 64 -> 32 elu, 32 -> 1
    float acc = bn2[lane];
    for (int i = 0; i < 64; ++i) acc += H1[i] * Wn2[lane * 64 + i];
    const float h2 = (acc > 0.0f) ? acc : (__expf(acc) - 1.0f);
    float p = h2 * Wn3[lane];
#pragma unroll
    for (int o = 16; o > 0; o >>= 1) p += __shfl_xor(p, o, 32);
    if (lane == 0) out[(long long)NN * 128 + n] = p + bn3[0];
  }
}

extern "C" void kernel_launch(void* const* d_in, const int* in_sizes, int n_in,
                              void* d_out, int out_size, void* d_ws, size_t ws_size,
                              hipStream_t stream){
  (void)in_sizes; (void)n_in; (void)out_size; (void)ws_size;
  const float*     x_in  = (const float*)d_in[0];
  const long long* eidx  = (const long long*)d_in[1];   // int64 [1,2,E]: src then dst
  const float*     eattr = (const float*)d_in[2];
  const float* bn0g=(const float*)d_in[3],  *bn0b=(const float*)d_in[4];
  const float* dummy=(const float*)d_in[5];
  const float* Wps=(const float*)d_in[6],   *bps=(const float*)d_in[7];
  const float* encg=(const float*)d_in[8],  *encbv=(const float*)d_in[9];
  const float* We1=(const float*)d_in[10],  *be1=(const float*)d_in[11];
  const float* We2=(const float*)d_in[12],  *be2=(const float*)d_in[13];
  const float* We3=(const float*)d_in[14],  *be3=(const float*)d_in[15];
  const float* Wc1=(const float*)d_in[16],  *bc1=(const float*)d_in[17];
  const float* Wc2=(const float*)d_in[18],  *bc2=(const float*)d_in[19];
  const float* Wa1=(const float*)d_in[20],  *ba1=(const float*)d_in[21];
  const float* Wa2=(const float*)d_in[22],  *ba2=(const float*)d_in[23];
  const float* Wa3=(const float*)d_in[24],  *ba3=(const float*)d_in[25];
  const float* Wb1=(const float*)d_in[26],  *bb1=(const float*)d_in[27];
  const float* Wb2=(const float*)d_in[28],  *bb2=(const float*)d_in[29];
  const float* Wb3=(const float*)d_in[30],  *bb3=(const float*)d_in[31];
  const float* bn1g=(const float*)d_in[32], *bn1b=(const float*)d_in[33];
  const float* bn2g=(const float*)d_in[34], *bn2b=(const float*)d_in[35];
  const float* Wg =(const float*)d_in[36],  *bg =(const float*)d_in[37];
  const float* Wn1=(const float*)d_in[38],  *bp1=(const float*)d_in[39];
  const float* Wn2=(const float*)d_in[40],  *bp2=(const float*)d_in[41];
  const float* Wn3=(const float*)d_in[42],  *bp3=(const float*)d_in[43];

  char* ws = (char*)d_ws;
  size_t off = 0;
  auto take = [&](size_t bytes) -> void* {
    void* p = ws + off;
    off += (bytes + 255) & ~(size_t)255;
    return p;
  };
  float*          xlnf  = (float*)take((size_t)NN * 32 * 4);
  unsigned short* xlnb  = (unsigned short*)take((size_t)NN * 32 * 2);
  unsigned short* x1b   = (unsigned short*)take((size_t)NN * 32 * 2);
  float*          efsum = (float*)take((size_t)NN * 64 * 4);
  float*          x1sum = (float*)take((size_t)NN * 32 * 4);
  float*          x2sum = (float*)take((size_t)NN * 64 * 4);
  float*          deg   = (float*)take((size_t)NN * 4);
  unsigned short* eencb = (unsigned short*)take((size_t)NE * 64 * 2);
  // pre-packed bf16 weights (K padded to multiple of 32)
  unsigned short* We1p = (unsigned short*)take((size_t)64  * 32  * 2);
  unsigned short* We2p = (unsigned short*)take((size_t)128 * 64  * 2);
  unsigned short* We3p = (unsigned short*)take((size_t)64  * 128 * 2);
  unsigned short* Wa1p = (unsigned short*)take((size_t)32  * 128 * 2);
  unsigned short* Wa2p = (unsigned short*)take((size_t)32  * 32  * 2);
  unsigned short* Wa3p = (unsigned short*)take((size_t)32  * 32  * 2);
  unsigned short* Wb1p = (unsigned short*)take((size_t)64  * 128 * 2);
  unsigned short* Wb2p = (unsigned short*)take((size_t)64  * 64  * 2);
  unsigned short* Wb3p = (unsigned short*)take((size_t)64  * 64  * 2);

  hipMemsetAsync(efsum, 0, (size_t)NN * 64 * 4, stream);
  hipMemsetAsync(x1sum, 0, (size_t)NN * 32 * 4, stream);
  hipMemsetAsync(x2sum, 0, (size_t)NN * 64 * 4, stream);
  hipMemsetAsync(deg,   0, (size_t)NN * 4,      stream);

  auto pack = [&](const float* W, unsigned short* out, int dout, int din, int dinPad){
    const int total = dout * dinPad;
    pack_w_kernel<<<(total + 255) / 256, 256, 0, stream>>>(W, out, dout, din, dinPad);
  };
  pack(We1, We1p, 64, 16, 32);
  pack(We2, We2p, 128, 64, 64);
  pack(We3, We3p, 64, 128, 128);
  pack(Wa1, Wa1p, 32, 128, 128);
  pack(Wa2, Wa2p, 32, 32, 32);
  pack(Wa3, Wa3p, 32, 32, 32);
  pack(Wb1, Wb1p, 64, 128, 128);
  pack(Wb2, Wb2p, 64, 64, 64);
  pack(Wb3, Wb3p, 64, 64, 64);

  const int nodeBlocks = (NN + 7) / 8;
  const int tileBlocks = ((NE + 15) / 16 + 3) / 4;

  node_pre_kernel<<<nodeBlocks, 256, 0, stream>>>(x_in, dummy, bn0g, bn0b, xlnf, xlnb);
  edge_enc_kernel<<<tileBlocks, 128, 0, stream>>>(eattr, eidx, encg, encbv,
                                                  We1p, be1, We2p, be2, We3p, be3,
                                                  Wc1, bc1, Wc2, bc2,
                                                  eencb, efsum, deg);
  conv_kernel<32><<<tileBlocks, 128, 0, stream>>>(xlnb, eencb, eidx,
                                                  Wa1p, ba1, Wa2p, ba2, Wa3p, ba3, x1sum);
  node_mid_kernel<<<nodeBlocks, 256, 0, stream>>>(x1sum, deg, bn1g, bn1b, x1b);
  conv_kernel<64><<<tileBlocks, 128, 0, stream>>>(x1b, eencb, eidx,
                                                  Wb1p, bb1, Wb2p, bb2, Wb3p, bb3, x2sum);
  node_final_kernel<<<nodeBlocks, 256, 0, stream>>>(x2sum, deg, efsum, xlnf,
                                                    bn2g, bn2b, Wps, bps, Wg, bg,
                                                    Wn1, bp1, Wn2, bp2, Wn3, bp3,
                                                    (float*)d_out);
}
